// LRPViewOfRelationNetwork_20521353741104
// MI455X (gfx1250) — matrix operand order repaired and checked by reading, not compiled, not run on hardware
//
#include <hip/hip_runtime.h>
#include <hip/hip_bf16.h>

typedef __attribute__((ext_vector_type(16))) __bf16 v16bf;
typedef __attribute__((ext_vector_type(8)))  float  v8f;
typedef __attribute__((ext_vector_type(4)))  unsigned int u32x4;
typedef __attribute__((ext_vector_type(8)))  int i32x8;
typedef __attribute__((ext_vector_type(4)))  int i32x4;

#define Bn   32
#define CH   24
#define MH   256
#define NCLS 28

// ---------------------------------------------------------------------------
// Generic stride-2 SAME 3x3 conv + bias + relu (tiny fraction of total work)
// ---------------------------------------------------------------------------
__global__ void rn_conv_kernel(const float* __restrict__ in, const float* __restrict__ w,
                               const float* __restrict__ bias, float* __restrict__ out,
                               int CI, int CO, int HI, int WI, int HO, int WO) {
    int idx = blockIdx.x * blockDim.x + threadIdx.x;
    int total = Bn * CO * HO * WO;
    if (idx >= total) return;
    int ox = idx % WO; int t = idx / WO;
    int oy = t % HO;   t /= HO;
    int co = t % CO;   int b = t / CO;
    float acc = bias[co];
    for (int ci = 0; ci < CI; ++ci) {
        const float* ip = in + ((size_t)(b * CI + ci) * HI) * WI;
        const float* wp = w + ((size_t)(co * CI + ci) * 9);
        #pragma unroll
        for (int ky = 0; ky < 3; ++ky) {
            int iy = oy * 2 + ky;           // SAME pad: pad_lo = 0, pad_hi = 1
            if (iy >= HI) continue;
            #pragma unroll
            for (int kx = 0; kx < 3; ++kx) {
                int ix = ox * 2 + kx;
                if (ix >= WI) continue;
                acc += ip[iy * WI + ix] * wp[ky * 3 + kx];
            }
        }
    }
    out[idx] = fmaxf(acc, 0.0f);
}

// ---------------------------------------------------------------------------
// A = conv_tr @ Wa, Bm = conv_tr @ Wb  (one block per (b, pair))
// ---------------------------------------------------------------------------
__global__ void rn_ab_kernel(const float* __restrict__ h4, const float* __restrict__ coords,
                             const float* __restrict__ gw1,
                             float* __restrict__ A, float* __restrict__ Bm) {
    int bp = blockIdx.x;              // b*64 + p
    int b = bp >> 6, p = bp & 63;
    int y = p >> 3, x = p & 7;
    int n = threadIdx.x;              // 0..255
    __shared__ float feat[26];
    if (n < CH)       feat[n] = h4[(((size_t)b * CH + n) * 8 + y) * 8 + x];
    else if (n < 26)  feat[n] = coords[(n - CH) * 64 + y * 8 + x];
    __syncthreads();
    float fa = 0.0f, fb = 0.0f;
    #pragma unroll
    for (int c = 0; c < 26; ++c) {
        float v = feat[c];
        fa += v * gw1[c * MH + n];
        fb += v * gw1[(26 + c) * MH + n];
    }
    A [(size_t)bp * MH + n] = fa;
    Bm[(size_t)bp * MH + n] = fb;
}

// ---------------------------------------------------------------------------
// Q = q @ Wq + gb1 ; also zero the gs accumulator for this launch
// ---------------------------------------------------------------------------
__global__ void rn_q_kernel(const float* __restrict__ q, const float* __restrict__ gw1,
                            const float* __restrict__ gb1,
                            float* __restrict__ Q, float* __restrict__ gs) {
    int b = blockIdx.x, n = threadIdx.x;
    float s = gb1[n];
    for (int k = 0; k < 128; ++k)
        s += q[b * 128 + k] * gw1[(52 + k) * MH + n];
    Q [b * MH + n] = s;
    gs[b * MH + n] = 0.0f;
}

// ---------------------------------------------------------------------------
// Transpose gw2/gw3/gw4 (K x N, fp32) -> Wt[layer][n][k] (bf16, contiguous K)
// ---------------------------------------------------------------------------
__global__ void rn_wt_kernel(const float* __restrict__ gw2, const float* __restrict__ gw3,
                             const float* __restrict__ gw4, __bf16* __restrict__ Wt) {
    int l = blockIdx.y;
    int n = blockIdx.x;
    int k = threadIdx.x;
    const float* src = (l == 0) ? gw2 : ((l == 1) ? gw3 : gw4);
    Wt[((size_t)l * MH + n) * MH + k] = (__bf16)src[(size_t)k * MH + n];
}

// ---------------------------------------------------------------------------
// TDM: load one 256-row x 32-elem bf16 weight K-panel into LDS.
// D# per ISA 8.3/8.4: 2D tile, data_size=2B, tile_dim0=32, tile_dim1=256,
// tensor_dim0_stride=256. Issued by one wave; tracked with TENSORcnt.
// (6-arg builtin variant: g0, g1, g2, g3, g4, cpol)
// ---------------------------------------------------------------------------
__device__ __forceinline__ void tdm_panel_load(const __bf16* gsrc, unsigned int lds_off) {
    unsigned long long ga = (unsigned long long)(size_t)gsrc;
    u32x4 g0;
    g0[0] = 1u;                                           // count=1, no gather
    g0[1] = lds_off;                                      // D#.lds_addr
    g0[2] = (unsigned int)ga;                             // global_addr[31:0]
    g0[3] = (unsigned int)((ga >> 32) & 0x01FFFFFFu)      // global_addr[56:32]
          | (2u << 30);                                   // type=2 ("image")
    i32x8 g1;
    g1[0] = (int)(1u << 16);                              // data_size=1 -> 2 bytes
    g1[1] = (int)((unsigned)MH << 16);                    // tensor_dim0 = 256 (lo16)
    g1[2] = (int)((unsigned)MH << 16);                    // dim0 hi=0 | tensor_dim1=256 lo16
    g1[3] = (int)(32u << 16);                             // dim1 hi=0 | tile_dim0=32
    g1[4] = MH;                                           // tile_dim1=256 | tile_dim2=0
    g1[5] = MH;                                           // tensor_dim0_stride = 256
    g1[6] = 0;
    g1[7] = 0;                                            // dim1_stride unused (2D)
    i32x4 z4 = {};                                        // groups 2/3 unused (2D tile)
    i32x8 z8 = {};
    __builtin_amdgcn_tensor_load_to_lds(g0, g1, z4, z4, z8, 0);
}

// ---------------------------------------------------------------------------
// Fused relation core: build x=relu(A[j]+Bm[i]+Q), 3x (WMMA 256x256 + relu),
// row-sum reduction. Weight K-panels streamed by the Tensor Data Mover with
// double buffering (panel p+1 in flight while panel p feeds the WMMAs).
// Block: 256 threads = 8 waves. 32 rows/block, grid = 32 batches * 128 blocks.
// ---------------------------------------------------------------------------
__global__ __launch_bounds__(256) void rn_relation_kernel(
    const float* __restrict__ A, const float* __restrict__ Bm, const float* __restrict__ Q,
    const __bf16* __restrict__ Wt,
    const float* __restrict__ gb2, const float* __restrict__ gb3, const float* __restrict__ gb4,
    float* __restrict__ gs) {

    __shared__ __bf16 Xbuf[2][32][MH];   // 32 KB ping-pong activations
    __shared__ __bf16 Wp[2][MH][32];     // 2 x 16 KB TDM-fed weight panels
    __shared__ float  gsum[MH];          // 1 KB per-block column sums

    const int tid  = threadIdx.x;
    const int lane = tid & 31;
    const int wave = tid >> 5;
    const int b    = blockIdx.x >> 7;    // 128 row-blocks per batch
    const int r0   = (blockIdx.x & 127) * 32;

    gsum[tid] = 0.0f;

    // ----- layer-0 input: x = relu(A[b,j] + Bm[b,i] + Q[b]) -> bf16 LDS -----
    const float* Ab = A  + (size_t)b * 64 * MH;
    const float* Bb = Bm + (size_t)b * 64 * MH;
    const float* Qb = Q  + (size_t)b * MH;
    for (int idx = tid; idx < 32 * MH; idx += 256) {
        int lr = idx >> 8;
        int n  = idx & 255;
        int r  = r0 + lr;
        int i  = r >> 6;
        int j  = r & 63;
        float v = Ab[j * MH + n] + Bb[i * MH + n] + Qb[n];
        Xbuf[0][lr][n] = (__bf16)fmaxf(v, 0.0f);
    }

    // per-lane fragment geometry (ISA 7.12.2 16-bit A / B layouts, wave32)
    const int m0    = (wave & 1) * 16;
    const int cg    = wave >> 1;
    const int arow  = m0 + (lane & 15);
    const int ksub  = (lane & 16) >> 1;   // 0 or 8  (A-matrix K half)
    const int koffB = (lane & 16);        // 0 or 16 (unused; B also uses ksub)
    const int ncol  = (lane & 15);
    const int mbase = m0 + ((lane & 16) >> 1);

    // 24 weight panels total: panel p = (layer p/8, K-block (p%8)*32)
    const unsigned int wp_lds[2] = {
        (unsigned int)(size_t)&Wp[0][0][0],
        (unsigned int)(size_t)&Wp[1][0][0]
    };
    // prologue: TDM fetch of panel 0 into Wp[0]
    if (wave == 0) tdm_panel_load(Wt, wp_lds[0]);

    const float* biases[3] = {gb2, gb3, gb4};
    int cur = 0;
    for (int layer = 0; layer < 3; ++layer) {
        const float* bias = biases[layer];
        v8f acc[4];
        #pragma unroll
        for (int t = 0; t < 4; ++t) acc[t] = v8f{};

        for (int i = 0; i < 8; ++i) {
            const int p  = layer * 8 + i;
            const int pb = p & 1;
            __syncthreads();   // next DMA target (Wp[pb^1]) no longer being read
            if (wave == 0) {
                if (p + 1 < 24) {
                    const __bf16* nsrc = Wt + (((size_t)(p + 1) >> 3) << 16)
                                            + (((p + 1) & 7) << 5);
                    tdm_panel_load(nsrc, wp_lds[pb ^ 1]);
                    __builtin_amdgcn_s_wait_tensorcnt(1);  // panel p landed (in-order)
                } else {
                    __builtin_amdgcn_s_wait_tensorcnt(0);
                }
            }
            __syncthreads();   // panel p visible to all waves

            const int kb = i * 32;
            // A fragment: lanes 0-15 row M=lane, K {kb+0..7, kb+16..23};
            //             lanes 16-31 same rows, K {kb+8..15, kb+24..31}
            v16bf af;
            {
                const __bf16* xr = &Xbuf[cur][arow][kb + ksub];
                #pragma unroll
                for (int e = 0; e < 8; ++e) { af[e] = xr[e]; af[8 + e] = xr[16 + e]; }
            }
            #pragma unroll
            for (int t = 0; t < 4; ++t) {
                const int n0 = (cg * 4 + t) * 16;
                // B fragment: lane -> col N=n0+(lane&15), K half by lane[4]
                v16bf bfrag;
                const __bf16* wr = &Wp[pb][n0 + ncol][ksub];
                #pragma unroll
                for (int e = 0; e < 8; ++e) { bfrag[e] = wr[e]; bfrag[8 + e] = wr[16 + e]; }
                acc[t] = __builtin_amdgcn_wmma_f32_16x16x32_bf16(
                    false, af, false, bfrag, (short)0, acc[t], false, false);
            }
        }

        // epilogue: bias + relu; middle layers -> ping-pong LDS, last -> reduce
        const int nxt = cur ^ 1;
        if (layer < 2) {
            #pragma unroll
            for (int t = 0; t < 4; ++t) {
                int n = (cg * 4 + t) * 16 + ncol;
                float bv = bias[n];
                #pragma unroll
                for (int r = 0; r < 8; ++r)
                    Xbuf[nxt][mbase + r][n] = (__bf16)fmaxf(acc[t][r] + bv, 0.0f);
            }
        } else {
            #pragma unroll
            for (int t = 0; t < 4; ++t) {
                int n = (cg * 4 + t) * 16 + ncol;
                float bv = bias[n];
                float s = 0.0f;
                #pragma unroll
                for (int r = 0; r < 8; ++r) s += fmaxf(acc[t][r] + bv, 0.0f);
                atomicAdd(&gsum[n], s);   // ds_add_f32
            }
        }
        cur = nxt;
    }

    __syncthreads();
    atomicAdd(&gs[(size_t)b * MH + tid], gsum[tid]);
}

// ---------------------------------------------------------------------------
// Head: f = relu(gs@fw1+fb1); f = relu(f@fw2+fb2); out = f@fw3+fb3
// ---------------------------------------------------------------------------
__global__ void rn_head_kernel(const float* __restrict__ gs,
                               const float* __restrict__ fw1, const float* __restrict__ fb1,
                               const float* __restrict__ fw2, const float* __restrict__ fb2,
                               const float* __restrict__ fw3, const float* __restrict__ fb3,
                               float* __restrict__ out) {
    __shared__ float s0[MH], s1[MH];
    int b = blockIdx.x, n = threadIdx.x;
    s0[n] = gs[b * MH + n];
    __syncthreads();
    float acc = fb1[n];
    for (int k = 0; k < MH; ++k) acc += s0[k] * fw1[k * MH + n];
    s1[n] = fmaxf(acc, 0.0f);
    __syncthreads();
    acc = fb2[n];
    for (int k = 0; k < MH; ++k) acc += s1[k] * fw2[k * MH + n];
    __syncthreads();
    s0[n] = fmaxf(acc, 0.0f);
    __syncthreads();
    if (n < NCLS) {
        float o = fb3[n];
        for (int k = 0; k < MH; ++k) o += s0[k] * fw3[k * NCLS + n];
        out[b * NCLS + n] = o;
    }
}

// ---------------------------------------------------------------------------
extern "C" void kernel_launch(void* const* d_in, const int* in_sizes, int n_in,
                              void* d_out, int out_size, void* d_ws, size_t ws_size,
                              hipStream_t stream) {
    (void)in_sizes; (void)n_in; (void)out_size; (void)ws_size;
    const float* image = (const float*)d_in[0];
    const float* lstm  = (const float*)d_in[1];
    const float* coords= (const float*)d_in[2];
    const float* cw1 = (const float*)d_in[3];  const float* cb1 = (const float*)d_in[4];
    const float* cw2 = (const float*)d_in[5];  const float* cb2 = (const float*)d_in[6];
    const float* cw3 = (const float*)d_in[7];  const float* cb3 = (const float*)d_in[8];
    const float* cw4 = (const float*)d_in[9];  const float* cb4 = (const float*)d_in[10];
    const float* gw1 = (const float*)d_in[11]; const float* gb1 = (const float*)d_in[12];
    const float* gw2 = (const float*)d_in[13]; const float* gb2 = (const float*)d_in[14];
    const float* gw3 = (const float*)d_in[15]; const float* gb3 = (const float*)d_in[16];
    const float* gw4 = (const float*)d_in[17]; const float* gb4 = (const float*)d_in[18];
    const float* fw1 = (const float*)d_in[19]; const float* fb1 = (const float*)d_in[20];
    const float* fw2 = (const float*)d_in[21]; const float* fb2 = (const float*)d_in[22];
    const float* fw3 = (const float*)d_in[23]; const float* fb3 = (const float*)d_in[24];
    float* out = (float*)d_out;

    char* ws = (char*)d_ws;
    size_t off = 0;
    float* h1 = (float*)(ws + off); off += (size_t)Bn * CH * 64 * 64 * 4;
    float* h2 = (float*)(ws + off); off += (size_t)Bn * CH * 32 * 32 * 4;
    float* h3 = (float*)(ws + off); off += (size_t)Bn * CH * 16 * 16 * 4;
    float* h4 = (float*)(ws + off); off += (size_t)Bn * CH * 8 * 8 * 4;
    float* Abuf = (float*)(ws + off); off += (size_t)Bn * 64 * MH * 4;
    float* Bbuf = (float*)(ws + off); off += (size_t)Bn * 64 * MH * 4;
    float* Qbuf = (float*)(ws + off); off += (size_t)Bn * MH * 4;
    float* gsb  = (float*)(ws + off); off += (size_t)Bn * MH * 4;
    __bf16* Wt  = (__bf16*)(ws + off); off += (size_t)3 * MH * MH * 2;

    // conv tower
    rn_conv_kernel<<<(Bn * CH * 64 * 64 + 255) / 256, 256, 0, stream>>>(image, cw1, cb1, h1, 3,  CH, 128, 128, 64, 64);
    rn_conv_kernel<<<(Bn * CH * 32 * 32 + 255) / 256, 256, 0, stream>>>(h1,    cw2, cb2, h2, CH, CH, 64,  64,  32, 32);
    rn_conv_kernel<<<(Bn * CH * 16 * 16 + 255) / 256, 256, 0, stream>>>(h2,    cw3, cb3, h3, CH, CH, 32,  32,  16, 16);
    rn_conv_kernel<<<(Bn * CH * 8  * 8  + 255) / 256, 256, 0, stream>>>(h3,    cw4, cb4, h4, CH, CH, 16,  16,  8,  8);

    // prep: weights -> bf16 transposed, A/Bm/Q projections, zero gs
    rn_wt_kernel<<<dim3(MH, 3), MH, 0, stream>>>(gw2, gw3, gw4, Wt);
    rn_ab_kernel<<<Bn * 64, MH, 0, stream>>>(h4, coords, gw1, Abuf, Bbuf);
    rn_q_kernel<<<Bn, MH, 0, stream>>>(lstm, gw1, gb1, Qbuf, gsb);

    // fused WMMA relation core: 32 rows/block, 128 blocks per batch
    rn_relation_kernel<<<Bn * 128, 256, 0, stream>>>(Abuf, Bbuf, Qbuf, Wt, gb2, gb3, gb4, gsb);

    // head
    rn_head_kernel<<<Bn, MH, 0, stream>>>(gsb, fw1, fb1, fw2, fb2, fw3, fb3, out);
}
